// QuantumMicroTransformer_7232724927093
// MI455X (gfx1250) — matrix operand order repaired and checked by reading, not compiled, not run on hardware
//
#include <hip/hip_runtime.h>
#include <hip/hip_bf16.h>
#include <math.h>

// ---------------------------------------------------------------------------
// Model constants (from the reference)
// ---------------------------------------------------------------------------
#define VSZ   100000
#define H     2048
#define NH    64
#define LAY   2
#define SEQ   512
#define BAT   2
#define HD    32          // H / NH
#define MROWS (BAT*SEQ)   // 1024

typedef __attribute__((ext_vector_type(16))) __bf16 v16bf;
typedef __attribute__((ext_vector_type(8)))  float  v8f;

union Frag { v16bf v; unsigned int u[8]; };

__device__ __forceinline__ unsigned int bf16_1(float f) {
    unsigned int u = __float_as_uint(f);
    u += 0x7FFFu + ((u >> 16) & 1u);      // round-to-nearest-even
    return u >> 16;
}
__device__ __forceinline__ unsigned int pack_bf16(float a, float b) {
    return bf16_1(a) | (bf16_1(b) << 16);
}

// ---------------------------------------------------------------------------
// Kernel 1: embedding gather * sqrt(H) + positional encoding (indexed by batch,
// faithful to the reference's pe[:x.shape[0]] quirk). Writes f32 + bf16 copies.
// ---------------------------------------------------------------------------
__global__ void embed_pe_kernel(const int* __restrict__ src,
                                const float* __restrict__ emb,
                                float* __restrict__ xf,
                                unsigned short* __restrict__ xbf) {
    size_t idx = (size_t)blockIdx.x * blockDim.x + threadIdx.x;
    if (idx >= (size_t)MROWS * H) return;
    int hcol = (int)(idx % H);
    int bs   = (int)(idx / H);          // b*SEQ + s
    int bpos = bs / SEQ;                // batch index used as "position"
    int tok  = src[bs];
    float v = emb[(size_t)tok * H + hcol] * 45.254833995939045f; // sqrt(2048)
    int i2 = hcol >> 1;
    float div = __expf((float)(2 * i2) * (-9.210340371976184f / (float)H)); // -ln(1e4)/H
    float ang = (float)bpos * div;
    float pe = (hcol & 1) ? __cosf(ang) : __sinf(ang);
    v += pe;
    xf[idx]  = v;
    xbf[idx] = (unsigned short)bf16_1(v);
}

// ---------------------------------------------------------------------------
// Kernel 2: generic GEMM  C[M,N] = A_bf16[M,K] @ W_f32[K,N] + bias
// N, K are template constants so all strided-load offsets become immediates.
// Block: 256 threads = 8 waves; wave tile 16(M)x64(N); block tile 64x128.
// Grid: (M/64, (N+127)/128)  -- M varies FASTEST so consecutive blocks share
// the same 1MB weight slab and hit L2 instead of re-streaming HBM.
// Writes f32 and/or bf16 output.
// ---------------------------------------------------------------------------
template <int N, int K>
__global__ void gemm_bf16_kernel(const unsigned short* __restrict__ A,
                                 const float* __restrict__ W,
                                 const float* __restrict__ bias,
                                 float* __restrict__ outF,
                                 unsigned short* __restrict__ outBF,
                                 int M) {
    constexpr bool EDGE = (N % 128) != 0;
    int lane = threadIdx.x & 31;
    int wave = threadIdx.x >> 5;
    int wm = wave & 3, wn = wave >> 2;
    int m0 = blockIdx.x * 64 + wm * 16;   // M-block index varies fastest
    int n0 = blockIdx.y * 128 + wn * 64;
    if (m0 >= M) return;
    int l15 = lane & 15;
    bool hi = (lane >= 16);

    // A fragment base: row m0+l15; contiguous K-half (16 per lane half).
    const unsigned short* aptr = A + (size_t)(m0 + l15) * K + (hi ? 16 : 0);

    // B fragment bases: one 64-bit base per 16-col subtile; all further
    // addressing is compile-time immediate offsets (j2 * N elements).
    int ncol[4];
    const float* bptr[4];
#pragma unroll
    for (int s = 0; s < 4; ++s) {
        int n = n0 + s * 16 + l15;
        ncol[s] = n;
        int nc = EDGE ? (n < N ? n : (N - 1)) : n;   // clamp, no exec branches
        bptr[s] = W + (size_t)(hi ? 16 : 0) * N + nc;
    }

    v8f zero = {0.f,0.f,0.f,0.f,0.f,0.f,0.f,0.f};
    v8f acc[4] = {zero, zero, zero, zero};

    for (int kk = 0; kk < K; kk += 32) {
        Frag fa;
#pragma unroll
        for (int j = 0; j < 8; ++j)
            fa.u[j] = *(const unsigned int*)(aptr + 2 * j);
        aptr += 32;
#pragma unroll
        for (int s = 0; s < 4; ++s) {
            Frag fb;
#pragma unroll
            for (int j = 0; j < 8; ++j) {
                float f0 = bptr[s][(size_t)(2 * j) * N];       // imm offsets
                float f1 = bptr[s][(size_t)(2 * j + 1) * N];
                fb.u[j] = pack_bf16(f0, f1);
            }
            bptr[s] += (size_t)32 * N;
            acc[s] = __builtin_amdgcn_wmma_f32_16x16x32_bf16(
                false, fa.v, false, fb.v, (short)0, acc[s], false, false);
        }
    }

#pragma unroll
    for (int s = 0; s < 4; ++s) {
        int n = ncol[s];
        if (EDGE && n >= N) continue;
        float bv = bias ? bias[n] : 0.f;
#pragma unroll
        for (int j = 0; j < 8; ++j) {
            int m = m0 + j + (hi ? 8 : 0);
            float v = acc[s][j] + bv;
            if (outF)  outF[(size_t)m * N + n]  = v;
            if (outBF) outBF[(size_t)m * N + n] = (unsigned short)bf16_1(v);
        }
    }
}

// ---------------------------------------------------------------------------
// Kernel 3: attention for one (batch, head, 16-query tile).
// scores = q k^T / sqrt(HD); softmax; e=exp(a); a2=e/(sum+eps); cw=a2*sigmoid(a2)
// o = cw @ v. Grid = BAT*NH*(SEQ/16) = 4096 blocks of 256 threads.
// ---------------------------------------------------------------------------
__global__ void attn_kernel(const unsigned short* __restrict__ qb,
                            const unsigned short* __restrict__ kb,
                            const unsigned short* __restrict__ vb,
                            unsigned short* __restrict__ ob) {
    __shared__ float sc[16][SEQ];            // 32 KB
    __shared__ unsigned short cwl[16][SEQ];  // 16 KB

    int idx = blockIdx.x;
    int qt = idx & 31;           // SEQ/16 = 32 tiles
    int hh = (idx >> 5) & 63;    // head
    int bb = idx >> 11;          // batch
    int lane = threadIdx.x & 31;
    int wave = threadIdx.x >> 5;
    int l15 = lane & 15;
    bool hi = (lane >= 16);
    int m0 = qt * 16;

    // ---- Phase 1: scores tile 16 x 512, wave w handles keys [64w, 64w+64) ----
    {
        Frag fa;
        int khalf = hi ? 16 : 0;
        const unsigned short* qrow =
            qb + ((size_t)(bb * SEQ + m0 + l15)) * H + hh * HD + khalf;
#pragma unroll
        for (int j = 0; j < 8; ++j)
            fa.u[j] = *(const unsigned int*)(qrow + 2 * j);
        v8f zero = {0.f,0.f,0.f,0.f,0.f,0.f,0.f,0.f};
        v8f acc[4] = {zero, zero, zero, zero};
#pragma unroll
        for (int s = 0; s < 4; ++s) {
            int key = wave * 64 + s * 16 + l15;
            const unsigned short* krow =
                kb + ((size_t)(bb * SEQ + key)) * H + hh * HD + (hi ? 16 : 0);
            Frag fb;
#pragma unroll
            for (int j = 0; j < 8; ++j)
                fb.u[j] = *(const unsigned int*)(krow + 2 * j);
            acc[s] = __builtin_amdgcn_wmma_f32_16x16x32_bf16(
                false, fa.v, false, fb.v, (short)0, acc[s], false, false);
        }
        const float scale = 0.17677669529663687f;  // 1/sqrt(32)
#pragma unroll
        for (int s = 0; s < 4; ++s)
#pragma unroll
            for (int j = 0; j < 8; ++j)
                sc[j + (hi ? 8 : 0)][wave * 64 + s * 16 + l15] = acc[s][j] * scale;
    }
    __syncthreads();

    // ---- Phase 2: per-row softmax -> exp -> renorm(+eps) -> sigmoid gate ----
#pragma unroll
    for (int rr = 0; rr < 2; ++rr) {
        int r = wave * 2 + rr;
        float vals[16];
#pragma unroll
        for (int t = 0; t < 16; ++t) vals[t] = sc[r][lane * 16 + t];
        float mx = vals[0];
#pragma unroll
        for (int t = 1; t < 16; ++t) mx = fmaxf(mx, vals[t]);
        for (int off = 16; off >= 1; off >>= 1)
            mx = fmaxf(mx, __shfl_xor(mx, off, 32));
        float sum = 0.f;
#pragma unroll
        for (int t = 0; t < 16; ++t) { vals[t] = __expf(vals[t] - mx); sum += vals[t]; }
        for (int off = 16; off >= 1; off >>= 1) sum += __shfl_xor(sum, off, 32);
        float inv = 1.f / sum;
        float sum2 = 0.f;
#pragma unroll
        for (int t = 0; t < 16; ++t) { vals[t] = __expf(vals[t] * inv); sum2 += vals[t]; }
        for (int off = 16; off >= 1; off >>= 1) sum2 += __shfl_xor(sum2, off, 32);
        float inv2 = 1.f / (sum2 + 1e-6f);
#pragma unroll
        for (int t = 0; t < 16; ++t) {
            float a2 = vals[t] * inv2;
            float g  = a2 / (1.f + __expf(-a2));   // a2 * sigmoid(a2)
            cwl[r][lane * 16 + t] = (unsigned short)bf16_1(g);
        }
    }
    __syncthreads();

    // ---- Phase 3: o[16x32] = cw[16x512] @ v[512x32]; waves 0,1 do 16 cols each
    if (wave < 2) {
        int d = wave * 16 + l15;
        const unsigned short* arow = &cwl[l15][hi ? 16 : 0];
        // V base: one pointer, advanced by 32 rows; row offsets are immediates.
        const unsigned short* vbase =
            vb + ((size_t)(bb * SEQ + (hi ? 16 : 0))) * H + hh * HD + d;
        v8f acc = {0.f,0.f,0.f,0.f,0.f,0.f,0.f,0.f};
        for (int kk = 0; kk < SEQ; kk += 32) {
            Frag fa;
#pragma unroll
            for (int j = 0; j < 8; ++j)
                fa.u[j] = *(const unsigned int*)(arow + 2 * j);
            arow += 32;
            Frag fb;
#pragma unroll
            for (int j = 0; j < 8; ++j) {
                unsigned int lo   = vbase[(size_t)(2 * j) * H];     // imm offsets
                unsigned int hi16 = vbase[(size_t)(2 * j + 1) * H];
                fb.u[j] = lo | (hi16 << 16);
            }
            vbase += (size_t)32 * H;
            acc = __builtin_amdgcn_wmma_f32_16x16x32_bf16(
                false, fa.v, false, fb.v, (short)0, acc, false, false);
        }
#pragma unroll
        for (int j = 0; j < 8; ++j) {
            int mr = m0 + j + (hi ? 8 : 0);
            ob[((size_t)(bb * SEQ + mr)) * H + hh * HD + d] =
                (unsigned short)bf16_1(acc[j]);
        }
    }
}

// ---------------------------------------------------------------------------
// Kernel 4: residual add + sr_norm with Quake fast inverse sqrt.
// One block (256 threads) per row of 2048.
// ---------------------------------------------------------------------------
__global__ void add_norm_kernel(const float* __restrict__ t,
                                float* __restrict__ x,
                                unsigned short* __restrict__ xbf,
                                const float* __restrict__ w,
                                const float* __restrict__ bcoef) {
    __shared__ float red[256];
    int row = blockIdx.x;
    int tid = threadIdx.x;
    float loc[8];
    float s = 0.f;
#pragma unroll
    for (int e = 0; e < 8; ++e) {
        int c = tid + e * 256;
        loc[e] = t[(size_t)row * H + c] + x[(size_t)row * H + c];
        s += loc[e];
    }
    red[tid] = s; __syncthreads();
    for (int st = 128; st > 0; st >>= 1) {
        if (tid < st) red[tid] += red[tid + st];
        __syncthreads();
    }
    float mean = red[0] * (1.f / (float)H);
    __syncthreads();
    float sq = 0.f;
#pragma unroll
    for (int e = 0; e < 8; ++e) { float d = loc[e] - mean; sq += d * d; }
    red[tid] = sq; __syncthreads();
    for (int st = 128; st > 0; st >>= 1) {
        if (tid < st) red[tid] += red[tid + st];
        __syncthreads();
    }
    float var = red[0] * (1.f / (float)H);
    // Quake fast inverse sqrt (one Newton step), exact constant from reference
    float v2 = var + 1e-5f;
    int i = __float_as_int(v2);
    i = 1597463007 - (i >> 1);
    float y = __int_as_float(i);
    y = y * (1.5f - 0.5f * v2 * y * y);
#pragma unroll
    for (int e = 0; e < 8; ++e) {
        int c = tid + e * 256;
        float o = w[c] * ((loc[e] - mean) * y) + bcoef[c];
        x[(size_t)row * H + c]   = o;
        xbf[(size_t)row * H + c] = (unsigned short)bf16_1(o);
    }
}

// ---------------------------------------------------------------------------
// Launch
// ---------------------------------------------------------------------------
extern "C" void kernel_launch(void* const* d_in, const int* in_sizes, int n_in,
                              void* d_out, int out_size, void* d_ws, size_t ws_size,
                              hipStream_t stream) {
    const int*   src  = (const int*)  d_in[0];
    const float* emb  = (const float*)d_in[1];
    const float* Wq   = (const float*)d_in[2];
    const float* bq   = (const float*)d_in[3];
    const float* Wk   = (const float*)d_in[4];
    const float* bk   = (const float*)d_in[5];
    const float* Wv   = (const float*)d_in[6];
    const float* bv   = (const float*)d_in[7];
    const float* Wo   = (const float*)d_in[8];
    const float* bo   = (const float*)d_in[9];
    const float* nw   = (const float*)d_in[10];
    const float* nb   = (const float*)d_in[11];
    const float* Wout = (const float*)d_in[12];
    const float* bout = (const float*)d_in[13];
    float* out = (float*)d_out;

    // Workspace carve-up (~36 MB)
    char* p = (char*)d_ws;
    float*          xf  = (float*)p;          p += (size_t)MROWS * H * 4;
    unsigned short* xbf = (unsigned short*)p; p += (size_t)MROWS * H * 2;
    unsigned short* qbf = (unsigned short*)p; p += (size_t)MROWS * H * 2;
    unsigned short* kbf = (unsigned short*)p; p += (size_t)MROWS * H * 2;
    unsigned short* vbf = (unsigned short*)p; p += (size_t)MROWS * H * 2;
    unsigned short* obf = (unsigned short*)p; p += (size_t)MROWS * H * 2;
    float*          tf  = (float*)p;          p += (size_t)MROWS * H * 4;

    // 1. embedding + positional encoding
    {
        size_t tot = (size_t)MROWS * H;
        embed_pe_kernel<<<(unsigned)((tot + 255) / 256), 256, 0, stream>>>(
            src, emb, xf, xbf);
    }

    // M-blocks on grid.x (fastest-varying) so same-weight-slab blocks are
    // temporally adjacent -> weight slab stays in L2, HBM streams W once.
    dim3 gHH(MROWS / 64, (H + 127) / 128);          // N=2048 GEMMs
    dim3 gV(MROWS / 64, (VSZ + 127) / 128);         // N=100000 GEMM
    dim3 blk(256);

    for (int l = 0; l < LAY; ++l) {
        const float* wq = Wq + (size_t)l * H * H;
        const float* wk = Wk + (size_t)l * H * H;
        const float* wv = Wv + (size_t)l * H * H;
        const float* wo = Wo + (size_t)l * H * H;

        gemm_bf16_kernel<H, H><<<gHH, blk, 0, stream>>>(xbf, wq, bq + l * H,
                                                        nullptr, qbf, MROWS);
        gemm_bf16_kernel<H, H><<<gHH, blk, 0, stream>>>(xbf, wk, bk + l * H,
                                                        nullptr, kbf, MROWS);
        gemm_bf16_kernel<H, H><<<gHH, blk, 0, stream>>>(xbf, wv, bv + l * H,
                                                        nullptr, vbf, MROWS);

        attn_kernel<<<BAT * NH * (SEQ / 16), blk, 0, stream>>>(qbf, kbf, vbf, obf);

        gemm_bf16_kernel<H, H><<<gHH, blk, 0, stream>>>(obf, wo, bo + l * H,
                                                        tf, nullptr, MROWS);

        add_norm_kernel<<<MROWS, blk, 0, stream>>>(tf, xf, xbf,
                                                   nw + l * H, nb + l * H);
    }

    // final projection into d_out (f32)
    gemm_bf16_kernel<VSZ, H><<<gV, blk, 0, stream>>>(xbf, Wout, bout,
                                                     out, nullptr, MROWS);
}
